// Model_19224273617344
// MI455X (gfx1250) — compile-verified, matrix operand order checked by reading, not run on hardware
//
#include <hip/hip_runtime.h>

// ---------------------------------------------------------------- types ----
typedef __bf16 bf16;
typedef __attribute__((ext_vector_type(16))) __bf16 v16bf;
typedef __attribute__((ext_vector_type(8)))  __bf16 v8bf;
typedef __attribute__((ext_vector_type(8)))  float  v8f;

// ------------------------------------------------------------- constants ---
constexpr int kL   = 2;
constexpr int kH   = 2048;
constexpr int kNH  = 16;
constexpr int kHD  = 128;
constexpr int kFF  = 4096;
constexpr int kS   = 1024;
constexpr int kB   = 2;
constexpr int kNL  = 4;
constexpr int kM   = kB * kS;      // 2048 token rows
constexpr float kSCALING = 8.0f;   // 16.0 / 2.0
constexpr int kBLK = 64;

__device__ __constant__ float g_nf4[16] = {
    -1.0f, -0.6961928009986877f, -0.5250730514526367f, -0.39491748809814453f,
    -0.28444138169288635f, -0.18477343022823334f, -0.09105003625154495f, 0.0f,
    0.07958029955625534f, 0.16093020141124725f, 0.24611230194568634f,
    0.33791524171829224f, 0.44070982933044434f, 0.5626170039176941f,
    0.7229568362236023f, 1.0f};

// --------------------------------------------------------- WMMA helpers ----
// CDNA5 16-bit A/B fragment (16x32): lane&15 = row (M for A, N for B),
// lane>>4 = half.  Elements 0..7  <-> K = k0 + half*8 + e   (contiguous 8)
//           Elements 8..15 <-> K = k0 + 16 + half*8 + (e-8) (contiguous 8)
__device__ __forceinline__ void load_frag_bf16(v16bf& f, const bf16* row_k0,
                                               int half) {
  const v8bf lo = *reinterpret_cast<const v8bf*>(row_k0 + half * 8);
  const v8bf hi = *reinterpret_cast<const v8bf*>(row_k0 + 16 + half * 8);
#pragma unroll
  for (int i = 0; i < 8; ++i) {
    f[i]     = lo[i];
    f[i + 8] = hi[i];
  }
}

__device__ __forceinline__ void load_frag_f32(v16bf& f, const float* row_k0,
                                              int half) {
#pragma unroll
  for (int i = 0; i < 8; ++i) {
    f[i]     = (bf16)row_k0[half * 8 + i];
    f[i + 8] = (bf16)row_k0[16 + half * 8 + i];
  }
}

__device__ __forceinline__ v8f wmma_bf16(v16bf a, v16bf b, v8f c) {
  return __builtin_amdgcn_wmma_f32_16x16x32_bf16(
      /*neg_a=*/false, a, /*neg_b=*/false, b,
      /*c_mod=*/(short)0, c, /*reuse_a=*/false, /*reuse_b=*/false);
}

// ------------------------------------------------------------- kernels -----
// NF4 dequant: W[n][k] = LUT[codes[n][k]] * scales[n][k/64]  -> bf16
__global__ __launch_bounds__(256) void dequant_nf4_kernel(
    const int* __restrict__ codes, const float* __restrict__ scales,
    bf16* __restrict__ W, int total, int Kdim) {
  __shared__ float lut[16];
  if (threadIdx.x < 16) lut[threadIdx.x] = g_nf4[threadIdx.x];
  __syncthreads();
  const int sb = Kdim / kBLK;
  for (int i = blockIdx.x * 256 + threadIdx.x; i < total;
       i += gridDim.x * 256) {
    int n = i / Kdim;
    int k = i - n * Kdim;
    int c = codes[i] & 15;
    W[i] = (bf16)(lut[c] * scales[n * sb + (k >> 6)]);
  }
}

// C[M,N] = A[M,K] @ B[N,K]^T   (bf16 in, f32 out)
// block: 256 thr = 8 waves, each wave a 32x32 tile; block tile 64x128.
__global__ __launch_bounds__(256) void gemm_bf16_wmma_kernel(
    const bf16* __restrict__ A, const bf16* __restrict__ Bm,
    float* __restrict__ C, int Ndim, int Kdim) {
  const int lane = threadIdx.x & 31;
  const int wave = threadIdx.x >> 5;
  const int half = lane >> 4;
  const int r    = lane & 15;
  const int m0 = blockIdx.y * 64 + (wave >> 2) * 32;
  const int n0 = blockIdx.x * 128 + (wave & 3) * 32;

  v8f acc00 = {}, acc01 = {}, acc10 = {}, acc11 = {};
  const bf16* arow0 = A + (size_t)(m0 + r) * Kdim;
  const bf16* arow1 = A + (size_t)(m0 + 16 + r) * Kdim;
  const bf16* brow0 = Bm + (size_t)(n0 + r) * Kdim;
  const bf16* brow1 = Bm + (size_t)(n0 + 16 + r) * Kdim;

  for (int k0 = 0; k0 < Kdim; k0 += 32) {
    __builtin_prefetch(arow0 + k0 + 256, 0, 1);
    __builtin_prefetch(brow0 + k0 + 256, 0, 1);
    v16bf a0, a1, b0, b1;
    load_frag_bf16(a0, arow0 + k0, half);
    load_frag_bf16(a1, arow1 + k0, half);
    load_frag_bf16(b0, brow0 + k0, half);
    load_frag_bf16(b1, brow1 + k0, half);
    acc00 = wmma_bf16(a0, b0, acc00);
    acc01 = wmma_bf16(a0, b1, acc01);
    acc10 = wmma_bf16(a1, b0, acc10);
    acc11 = wmma_bf16(a1, b1, acc11);
  }
  // C/D layout: lane&15 = column, VGPR i -> row i + 8*half
#pragma unroll
  for (int i = 0; i < 8; ++i) {
    const int mr = i + 8 * half;
    C[(size_t)(m0 + mr) * Ndim + n0 + r]           = acc00[i];
    C[(size_t)(m0 + mr) * Ndim + n0 + 16 + r]      = acc01[i];
    C[(size_t)(m0 + 16 + mr) * Ndim + n0 + r]      = acc10[i];
    C[(size_t)(m0 + 16 + mr) * Ndim + n0 + 16 + r] = acc11[i];
  }
}

// x[m,:] = embed[ids[m],:]
__global__ __launch_bounds__(256) void embed_kernel(
    const int* __restrict__ ids, const float* __restrict__ emb,
    float* __restrict__ x) {
  const int m  = blockIdx.x;
  const int id = ids[m];
  for (int i = threadIdx.x; i < kH; i += 256)
    x[(size_t)m * kH + i] = emb[(size_t)id * kH + i];
}

__global__ __launch_bounds__(256) void lens_kernel(const int* __restrict__ am,
                                                   int* __restrict__ lens) {
  __shared__ int red[256];
  const int b = blockIdx.x;
  int s = 0;
  for (int i = threadIdx.x; i < kS; i += 256) s += am[b * kS + i];
  red[threadIdx.x] = s;
  __syncthreads();
  for (int o = 128; o > 0; o >>= 1) {
    if (threadIdx.x < o) red[threadIdx.x] += red[threadIdx.x + o];
    __syncthreads();
  }
  if (threadIdx.x == 0) lens[b] = red[0];
}

// LayerNorm over width kH; bf16 and/or f32 outputs (either may be null)
__global__ __launch_bounds__(256) void layernorm_kernel(
    const float* __restrict__ x, const float* __restrict__ g,
    const float* __restrict__ bb, bf16* __restrict__ obf,
    float* __restrict__ of) {
  __shared__ float red[256], red2[256];
  const int row = blockIdx.x;
  const float* xr = x + (size_t)row * kH;
  float s = 0.f, s2 = 0.f;
  for (int i = threadIdx.x; i < kH; i += 256) {
    float v = xr[i];
    s += v;
    s2 += v * v;
  }
  red[threadIdx.x] = s;
  red2[threadIdx.x] = s2;
  __syncthreads();
  for (int o = 128; o > 0; o >>= 1) {
    if (threadIdx.x < o) {
      red[threadIdx.x] += red[threadIdx.x + o];
      red2[threadIdx.x] += red2[threadIdx.x + o];
    }
    __syncthreads();
  }
  const float mu  = red[0] * (1.0f / kH);
  const float var = red2[0] * (1.0f / kH) - mu * mu;
  const float rs  = rsqrtf(var + 1e-5f);
  for (int i = threadIdx.x; i < kH; i += 256) {
    float v = (xr[i] - mu) * rs * g[i] + bb[i];
    if (obf) obf[(size_t)row * kH + i] = (bf16)v;
    if (of) of[(size_t)row * kH + i] = v;
  }
}

// t[m,0:2] = Xb[m,:] @ A[0:2,:]^T
__global__ __launch_bounds__(256) void lora_xa_kernel(
    const bf16* __restrict__ Xb, const float* __restrict__ Amat,
    float* __restrict__ T, int Kdim) {
  __shared__ float r0[256], r1[256];
  const int m = blockIdx.x;
  const bf16* xr = Xb + (size_t)m * Kdim;
  float t0 = 0.f, t1 = 0.f;
  for (int i = threadIdx.x; i < Kdim; i += 256) {
    float v = (float)xr[i];
    t0 += v * Amat[i];
    t1 += v * Amat[Kdim + i];
  }
  r0[threadIdx.x] = t0;
  r1[threadIdx.x] = t1;
  __syncthreads();
  for (int o = 128; o > 0; o >>= 1) {
    if (threadIdx.x < o) {
      r0[threadIdx.x] += r0[threadIdx.x + o];
      r1[threadIdx.x] += r1[threadIdx.x + o];
    }
    __syncthreads();
  }
  if (threadIdx.x == 0) {
    T[m * 2]     = r0[0];
    T[m * 2 + 1] = r1[0];
  }
}

// gemm_out += SCALING * t @ Bm^T ; optionally x_resid += gemm_out(+lora)
__global__ __launch_bounds__(256) void lora_add_kernel(
    float* __restrict__ gemm_out, const float* __restrict__ T,
    const float* __restrict__ Bm, float* __restrict__ x_resid, int Ndim,
    int total) {
  for (int i = blockIdx.x * 256 + threadIdx.x; i < total;
       i += gridDim.x * 256) {
    int m = i / Ndim;
    int n = i - m * Ndim;
    float l = kSCALING * (T[m * 2] * Bm[n * 2] + T[m * 2 + 1] * Bm[n * 2 + 1]);
    if (x_resid)
      x_resid[i] += gemm_out[i] + l;
    else
      gemm_out[i] += l;
  }
}

// Split qkv (B,S,NH,3*HD f32), rope q/k, write q/k (B,NH,S,HD) bf16 and
// v transposed (B,NH,HD,S) bf16.
__global__ __launch_bounds__(256) void rope_split_kernel(
    const float* __restrict__ qkv, bf16* __restrict__ qd,
    bf16* __restrict__ kd, bf16* __restrict__ vd) {
  const int idx = blockIdx.x * 256 + threadIdx.x;
  if (idx >= kB * kS * kNH * kHD) return;
  const int d = idx & (kHD - 1);
  const int h = (idx >> 7) & (kNH - 1);
  const int s = (idx >> 11) & (kS - 1);
  const int b = idx >> 21;
  const size_t base = ((size_t)(b * kS + s)) * (3 * kH) + (size_t)h * (3 * kHD);
  // inv = 10000^(-(d&63)/64) = exp(-(d&63) * ln(10000)/64)
  const float ang = (float)s * __expf(-(float)(d & 63) * 0.14391156516f);
  const float c = __cosf(ang), sn = __sinf(ang);
  const float qv = qkv[base + d];
  const float qo = (d < 64) ? -qkv[base + d + 64] : qkv[base + d - 64];
  const float kv = qkv[base + kHD + d];
  const float ko = (d < 64) ? -qkv[base + kHD + d + 64]
                            : qkv[base + kHD + d - 64];
  const size_t rm = (((size_t)(b * kNH + h)) * kS + s) * kHD + d;
  qd[rm] = (bf16)(qv * c + qo * sn);
  kd[rm] = (bf16)(kv * c + ko * sn);
  vd[(((size_t)(b * kNH + h)) * kHD + d) * kS + s] = (bf16)qkv[base + 2 * kHD + d];
}

// One block = (b, head, 16-row q tile).  Scores row-block in LDS,
// QK^T and PV via WMMA.  attn_out (B,S,H) bf16, already /rowsum.
__global__ __launch_bounds__(256) void attention_kernel(
    const bf16* __restrict__ qd, const bf16* __restrict__ kd,
    const bf16* __restrict__ vt, const int* __restrict__ lens,
    bf16* __restrict__ attn_out) {
  __shared__ float s_sc[16 * kS];       // 64 KB
  __shared__ float red[256];
  __shared__ float rowmax[16], rowsum[16];

  const int bid = blockIdx.x;
  const int qt = bid & 63;
  const int h  = (bid >> 6) & (kNH - 1);
  const int b  = bid >> 10;
  const int q0 = qt * 16;

  const int lane = threadIdx.x & 31;
  const int wave = threadIdx.x >> 5;
  const int half = lane >> 4;
  const int r    = lane & 15;

  const bf16* qbase = qd + ((size_t)(b * kNH + h)) * kS * kHD;
  const bf16* kbase = kd + ((size_t)(b * kNH + h)) * kS * kHD;
  const bf16* vbase = vt + ((size_t)(b * kNH + h)) * kHD * kS;
  const int len = lens[b];

  // ---- phase 1: scores = QK^T / sqrt(HD), masked, to LDS ----
  v16bf qa[4];
#pragma unroll
  for (int kk = 0; kk < 4; ++kk)
    load_frag_bf16(qa[kk], qbase + (size_t)(q0 + r) * kHD + kk * 32, half);

  const float scale = 0.08838834764831845f;  // 1/sqrt(128)
  for (int t = 0; t < 8; ++t) {
    const int n0 = (wave * 8 + t) * 16;
    v8f acc = {};
#pragma unroll
    for (int kk = 0; kk < 4; ++kk) {
      v16bf kb;
      load_frag_bf16(kb, kbase + (size_t)(n0 + r) * kHD + kk * 32, half);
      acc = wmma_bf16(qa[kk], kb, acc);
    }
#pragma unroll
    for (int i = 0; i < 8; ++i) {
      const int m   = i + 8 * half;
      const int col = n0 + r;
      float v = acc[i] * scale;
      if (!((col <= q0 + m) && (col < len))) v = -1e30f;
      s_sc[m * kS + col] = v;
    }
  }
  __syncthreads();

  // ---- phase 2: softmax over 1024 per row ----
  const int row = threadIdx.x & 15;
  const int chunk = threadIdx.x >> 4;  // 16 chunks of 64 columns
  float pm = -3.4e38f;
  for (int c = chunk * 64; c < chunk * 64 + 64; ++c)
    pm = fmaxf(pm, s_sc[row * kS + c]);
  red[threadIdx.x] = pm;
  __syncthreads();
  if (chunk == 0) {
    float mm = red[row];
    for (int c = 1; c < 16; ++c) mm = fmaxf(mm, red[c * 16 + row]);
    rowmax[row] = mm;
  }
  __syncthreads();
  const float rmax = rowmax[row];
  float ps = 0.f;
  for (int c = chunk * 64; c < chunk * 64 + 64; ++c) {
    float e = __expf(s_sc[row * kS + c] - rmax);
    s_sc[row * kS + c] = e;
    ps += e;
  }
  red[threadIdx.x] = ps;
  __syncthreads();
  if (chunk == 0) {
    float ss = red[row];
    for (int c = 1; c < 16; ++c) ss += red[c * 16 + row];
    rowsum[row] = ss;
  }
  __syncthreads();
  const float inv = 1.0f / rowsum[row];
  for (int c = chunk * 64; c < chunk * 64 + 64; ++c)
    s_sc[row * kS + c] *= inv;
  __syncthreads();

  // ---- phase 3: O = P @ V  (V stored transposed: row d contiguous in k) ----
  const int dbase = wave * 16;
  v8f o = {};
  for (int k0 = 0; k0 < kS; k0 += 32) {
    v16bf pa, vb;
    load_frag_f32(pa, s_sc + r * kS + k0, half);
    load_frag_bf16(vb, vbase + (size_t)(dbase + r) * kS + k0, half);
    o = wmma_bf16(pa, vb, o);
  }
#pragma unroll
  for (int i = 0; i < 8; ++i) {
    const int m = i + 8 * half;
    attn_out[((size_t)(b * kS + q0 + m)) * kH + h * kHD + dbase + r] =
        (bf16)o[i];
  }
}

// g = silu(u[:, :FF]) * u[:, FF:]
__global__ __launch_bounds__(256) void silu_mul_kernel(
    const float* __restrict__ u, bf16* __restrict__ g) {
  const int idx = blockIdx.x * 256 + threadIdx.x;
  if (idx >= kM * kFF) return;
  const int m = idx / kFF;
  const int f = idx - m * kFF;
  const float a = u[(size_t)m * (2 * kFF) + f];
  const float b = u[(size_t)m * (2 * kFF) + kFF + f];
  g[idx] = (bf16)((a / (1.0f + __expf(-a))) * b);
}

// logits + log_softmax + NLL, single block
__global__ __launch_bounds__(256) void loss_kernel(
    const float* __restrict__ xf, const float* __restrict__ wc,
    const int* __restrict__ lens, const int* __restrict__ labels,
    float* __restrict__ out) {
  __shared__ float red[256];
  __shared__ float lg[kB * kNL];
  for (int b = 0; b < kB; ++b) {
    const int eos = lens[b] - 1;
    const float* xr = xf + ((size_t)(b * kS + eos)) * kH;
    for (int c = 0; c < kNL; ++c) {
      float s = 0.f;
      for (int i = threadIdx.x; i < kH; i += 256)
        s += xr[i] * wc[(size_t)c * kH + i];
      red[threadIdx.x] = s;
      __syncthreads();
      for (int o = 128; o > 0; o >>= 1) {
        if (threadIdx.x < o) red[threadIdx.x] += red[threadIdx.x + o];
        __syncthreads();
      }
      if (threadIdx.x == 0) lg[b * kNL + c] = red[0];
      __syncthreads();
    }
  }
  if (threadIdx.x == 0) {
    float loss = 0.f;
    for (int b = 0; b < kB; ++b) {
      float mx = lg[b * kNL];
      for (int c = 1; c < kNL; ++c) mx = fmaxf(mx, lg[b * kNL + c]);
      float se = 0.f;
      for (int c = 0; c < kNL; ++c) se += __expf(lg[b * kNL + c] - mx);
      const float lse = mx + __logf(se);
      loss += -(lg[b * kNL + labels[b]] - lse);
    }
    out[0] = loss / kB;
    for (int i = 0; i < kB * kNL; ++i) out[1 + i] = lg[i];
  }
}

// --------------------------------------------------------------- launch ----
extern "C" void kernel_launch(void* const* d_in, const int* in_sizes, int n_in,
                              void* d_out, int out_size, void* d_ws,
                              size_t ws_size, hipStream_t stream) {
  (void)in_sizes; (void)n_in; (void)out_size; (void)ws_size;
  const int*   input_ids  = (const int*)d_in[0];
  const int*   attn_mask  = (const int*)d_in[1];
  const int*   labels     = (const int*)d_in[2];
  const float* embed      = (const float*)d_in[3];
  const int*   qkv_codes  = (const int*)d_in[4];
  const float* qkv_scales = (const float*)d_in[5];
  const float* qkv_A      = (const float*)d_in[6];
  const float* qkv_B      = (const float*)d_in[7];
  const int*   den_codes  = (const int*)d_in[8];
  const float* den_scales = (const float*)d_in[9];
  const float* den_A      = (const float*)d_in[10];
  const float* den_B      = (const float*)d_in[11];
  const int*   min_codes  = (const int*)d_in[12];
  const float* min_scales = (const float*)d_in[13];
  const float* min_A      = (const float*)d_in[14];
  const float* min_B      = (const float*)d_in[15];
  const int*   mout_codes = (const int*)d_in[16];
  const float* mout_scales= (const float*)d_in[17];
  const float* mout_A     = (const float*)d_in[18];
  const float* mout_B     = (const float*)d_in[19];
  const float* ln1_g = (const float*)d_in[20];
  const float* ln1_b = (const float*)d_in[21];
  const float* ln2_g = (const float*)d_in[22];
  const float* ln2_b = (const float*)d_in[23];
  const float* lnf_g = (const float*)d_in[24];
  const float* lnf_b = (const float*)d_in[25];
  const float* cls_w = (const float*)d_in[26];
  float* out = (float*)d_out;

  // ---- workspace carve ----
  char* p = (char*)d_ws;
  auto take = [&](size_t bytes) -> void* {
    void* q = (void*)p;
    p += (bytes + 255) & ~(size_t)255;
    return q;
  };
  float* x     = (float*)take((size_t)kM * kH * 4);
  bf16*  hbf   = (bf16*)take((size_t)kM * kFF * 2);   // LN out / gate out
  bf16*  attnb = (bf16*)take((size_t)kM * kH * 2);
  float* tmp   = (float*)take((size_t)kM * (2 * kFF) * 4);
  bf16*  wdq   = (bf16*)take((size_t)(2 * kFF) * kH * 2);
  bf16*  q_t   = (bf16*)take((size_t)kB * kNH * kS * kHD * 2);
  bf16*  k_t   = (bf16*)take((size_t)kB * kNH * kS * kHD * 2);
  bf16*  v_t   = (bf16*)take((size_t)kB * kNH * kHD * kS * 2);
  float* tvec  = (float*)take((size_t)kM * 2 * 4);
  int*   lensp = (int*)take(256);

  lens_kernel<<<kB, 256, 0, stream>>>(attn_mask, lensp);
  embed_kernel<<<kM, 256, 0, stream>>>(input_ids, embed, x);

  for (int l = 0; l < kL; ++l) {
    // ---------------- attention block ----------------
    layernorm_kernel<<<kM, 256, 0, stream>>>(x, ln1_g + l * kH, ln1_b + l * kH,
                                             hbf, (float*)nullptr);
    // qkv = h @ Wqkv^T + lora
    dequant_nf4_kernel<<<8192, 256, 0, stream>>>(
        qkv_codes + (size_t)l * 3 * kH * kH,
        qkv_scales + (size_t)l * 3 * kH * (kH / kBLK), wdq, 3 * kH * kH, kH);
    gemm_bf16_wmma_kernel<<<dim3(3 * kH / 128, kM / 64), 256, 0, stream>>>(
        hbf, wdq, tmp, 3 * kH, kH);
    lora_xa_kernel<<<kM, 256, 0, stream>>>(hbf, qkv_A + (size_t)l * 2 * kH,
                                           tvec, kH);
    lora_add_kernel<<<8192, 256, 0, stream>>>(
        tmp, tvec, qkv_B + (size_t)l * 3 * kH * 2, (float*)nullptr, 3 * kH,
        kM * 3 * kH);
    rope_split_kernel<<<(kB * kS * kNH * kHD) / 256, 256, 0, stream>>>(
        tmp, q_t, k_t, v_t);
    attention_kernel<<<kB * kNH * (kS / 16), 256, 0, stream>>>(
        q_t, k_t, v_t, lensp, attnb);
    // x += attn @ Wdense^T + lora
    dequant_nf4_kernel<<<8192, 256, 0, stream>>>(
        den_codes + (size_t)l * kH * kH,
        den_scales + (size_t)l * kH * (kH / kBLK), wdq, kH * kH, kH);
    gemm_bf16_wmma_kernel<<<dim3(kH / 128, kM / 64), 256, 0, stream>>>(
        attnb, wdq, tmp, kH, kH);
    lora_xa_kernel<<<kM, 256, 0, stream>>>(attnb, den_A + (size_t)l * 2 * kH,
                                           tvec, kH);
    lora_add_kernel<<<8192, 256, 0, stream>>>(
        tmp, tvec, den_B + (size_t)l * kH * 2, x, kH, kM * kH);

    // ---------------- MLP block ----------------
    layernorm_kernel<<<kM, 256, 0, stream>>>(x, ln2_g + l * kH, ln2_b + l * kH,
                                             hbf, (float*)nullptr);
    dequant_nf4_kernel<<<8192, 256, 0, stream>>>(
        min_codes + (size_t)l * 2 * kFF * kH,
        min_scales + (size_t)l * 2 * kFF * (kH / kBLK), wdq, 2 * kFF * kH, kH);
    gemm_bf16_wmma_kernel<<<dim3(2 * kFF / 128, kM / 64), 256, 0, stream>>>(
        hbf, wdq, tmp, 2 * kFF, kH);
    lora_xa_kernel<<<kM, 256, 0, stream>>>(hbf, min_A + (size_t)l * 2 * kH,
                                           tvec, kH);
    lora_add_kernel<<<8192, 256, 0, stream>>>(
        tmp, tvec, min_B + (size_t)l * 2 * kFF * 2, (float*)nullptr, 2 * kFF,
        kM * 2 * kFF);
    silu_mul_kernel<<<(kM * kFF) / 256, 256, 0, stream>>>(tmp, hbf);
    dequant_nf4_kernel<<<8192, 256, 0, stream>>>(
        mout_codes + (size_t)l * kH * kFF,
        mout_scales + (size_t)l * kH * (kFF / kBLK), wdq, kH * kFF, kFF);
    gemm_bf16_wmma_kernel<<<dim3(kH / 128, kM / 64), 256, 0, stream>>>(
        hbf, wdq, tmp, kH, kFF);
    lora_xa_kernel<<<kM, 256, 0, stream>>>(hbf, mout_A + (size_t)l * 2 * kFF,
                                           tvec, kFF);
    lora_add_kernel<<<8192, 256, 0, stream>>>(
        tmp, tvec, mout_B + (size_t)l * kH * 2, x, kH, kM * kH);
  }

  // final LN (f32 into tmp) -> classifier -> loss
  layernorm_kernel<<<kM, 256, 0, stream>>>(x, lnf_g, lnf_b, (bf16*)nullptr,
                                           tmp);
  loss_kernel<<<1, 256, 0, stream>>>(tmp, cls_w, lensp, labels, out);
}
